// OST_8950711845626
// MI455X (gfx1250) — compile-verified, matrix operand order checked by reading, not run on hardware
//
#include <hip/hip_runtime.h>
#include <hip/hip_bf16.h>
#include <stdint.h>

namespace {
constexpr int Bn = 256, Tn = 16, An = 400, Nn = 8, Dn = 512;
constexpr int KC     = Dn / 32;    // 16 K-chunks of 32
constexpr int APAIR  = An / 2;     // 200 descriptor pairs (16-wide N tiles)
constexpr int FRAG   = 32 * 16;    // halfs per k-chunk fragment (32 lanes x 16 halfs)
constexpr int TDM_ELEMS = KC * FRAG;  // 8192 halfs = 16KB per A fragment array
constexpr float INV_EPS = 10.0f;   // 1/EPS, EPS=0.1
constexpr float THRESH  = 0.01f;
constexpr int   MAX_ITER = 100;
}

typedef __attribute__((ext_vector_type(16))) __bf16 v16bf;
typedef __attribute__((ext_vector_type(8)))  float  v8f;
typedef __attribute__((ext_vector_type(4)))  unsigned int v4u;
typedef __attribute__((ext_vector_type(8)))  int v8i;
typedef __attribute__((ext_vector_type(4)))  int v4i;

// DPP8 selector immediates for lane^m within 8-lane groups: imm = sum((i^m)<<(3i))
#define DPP8_XOR1 14570689
#define DPP8_XOR2 11788826
#define DPP8_XOR4 6852524
// butterfly add with lane^m partner, pure-VALU (v_mov_b32 dpp8), no LDS pipe
#define XOR_ADD(x, IMM) \
  ((x) + __int_as_float(__builtin_amdgcn_mov_dpp8(__float_as_int(x), (IMM))))

// split fp32 into bf16 hi + bf16 lo (truncation; residual ~2^-16 relative)
__device__ __forceinline__ void bf16_split(float x, uint16_t& hi, uint16_t& lo) {
  union { float f; uint32_t u; } cv; cv.f = x;
  uint32_t hb = cv.u & 0xFFFF0000u;
  hi = (uint16_t)(hb >> 16);
  union { uint32_t u; float f; } hv; hv.u = hb;
  union { float f; uint32_t u; } rv; rv.f = x - hv.f;
  lo = (uint16_t)(rv.u >> 16);
}

// CDNA5 16-bit A/B fragment slot for element (row-or-col rc, k-offset-in-chunk ko):
// lanes 0-15: V0-3 hold K 0..7, V4-7 hold K 16..23; lanes 16-31: K 8..15 / 24..31.
__device__ __forceinline__ void frag_slot(int rc, int ko, int& lp, int& jj) {
  if (ko < 8)       { lp = rc;      jj = ko;      }
  else if (ko < 16) { lp = rc + 16; jj = ko - 8;  }
  else if (ko < 24) { lp = rc;      jj = ko - 8;  }
  else              { lp = rc + 16; jj = ko - 16; }
}

// TDM 1-D copy: 16KB contiguous global -> LDS (D# per ISA cdna5 ch.8)
__device__ __forceinline__ void tdm_load_1d(const void* gsrc, uint32_t lds_off) {
  uint64_t ga = (uint64_t)(uintptr_t)gsrc;
  v4u g0;
  g0[0] = 1u;                                  // count=1, user descriptor
  g0[1] = lds_off;                             // lds_addr (bytes)
  g0[2] = (uint32_t)ga;                        // global_addr[31:0]
  g0[3] = ((uint32_t)(ga >> 32) & 0x01FFFFFFu) // global_addr[56:32]
          | 0x80000000u;                       // type=2 ("image")
  v8i g1;
  g1[0] = 0x00010000;                // data_size=2B; no multicast/pad/iterate
  g1[1] = (int)(TDM_ELEMS << 16);    // tensor_dim0[15:0] -> bits[63:48]
  g1[2] = 0x00010000;                // tensor_dim1 = 1 -> bits[95:80]
  g1[3] = (int)(TDM_ELEMS << 16);    // tile_dim0 -> bits[127:112]
  g1[4] = 1;                         // tile_dim1 = 1
  g1[5] = (int)TDM_ELEMS;            // tensor_dim0_stride
  g1[6] = 0; g1[7] = 0;
  v4i z4 = {0, 0, 0, 0};
#if __clang_major__ >= 23
  v8i z8 = {0, 0, 0, 0, 0, 0, 0, 0};
  __builtin_amdgcn_tensor_load_to_lds(g0, g1, z4, z4, z8, 0);
#else
  __builtin_amdgcn_tensor_load_to_lds(g0, g1, z4, z4, 0);
#endif
}

// ---- normalize image rows, emit WMMA A-fragments (hi/lo) + row inv-norms -----
__global__ void ost_norm_pack_img(const float* __restrict__ imf,
                                  uint16_t* __restrict__ pk_hi,
                                  uint16_t* __restrict__ pk_lo,
                                  float* __restrict__ rn_img) {
  int wid  = (int)((blockIdx.x * blockDim.x + threadIdx.x) >> 5);
  int lane = (int)(threadIdx.x & 31);
  if (wid >= Bn * Tn) return;
  int b = wid >> 4, t = wid & 15;
  const float* row = imf + (size_t)wid * Dn;
  float x[16]; float ss = 0.f;
#pragma unroll
  for (int j = 0; j < 16; ++j) { x[j] = row[lane + 32 * j]; ss += x[j] * x[j]; }
#pragma unroll
  for (int m = 1; m < 32; m <<= 1) ss += __shfl_xor(ss, m, 32);
  float rn = rsqrtf(ss);
  if (lane == 0) rn_img[wid] = rn;
#pragma unroll
  for (int j = 0; j < 16; ++j) {
    int k = lane + 32 * j;
    uint16_t hi, lo; bf16_split(x[j] * rn, hi, lo);
    int kc = k >> 5, ko = k & 31, lp, jj;
    frag_slot(t, ko, lp, jj);
    size_t idx = (((size_t)b * KC + kc) * 32 + lp) * 16 + jj;
    pk_hi[idx] = hi; pk_lo[idx] = lo;
  }
}

// ---- normalize descriptor rows, emit WMMA B-fragments (hi/lo) + inv-norms ----
__global__ void ost_norm_pack_desc(const float* __restrict__ desc,
                                   uint16_t* __restrict__ pk_hi,
                                   uint16_t* __restrict__ pk_lo,
                                   float* __restrict__ rn_desc) {
  int wid  = (int)((blockIdx.x * blockDim.x + threadIdx.x) >> 5);
  int lane = (int)(threadIdx.x & 31);
  if (wid >= An * Nn) return;
  int a = wid >> 3, n = wid & 7;
  int apair = a >> 1, col = (a & 1) * 8 + n;  // 16-wide N tile = two descriptors
  const float* row = desc + (size_t)wid * Dn;
  float x[16]; float ss = 0.f;
#pragma unroll
  for (int j = 0; j < 16; ++j) { x[j] = row[lane + 32 * j]; ss += x[j] * x[j]; }
#pragma unroll
  for (int m = 1; m < 32; m <<= 1) ss += __shfl_xor(ss, m, 32);
  float rn = rsqrtf(ss);
  if (lane == 0) rn_desc[wid] = rn;
#pragma unroll
  for (int j = 0; j < 16; ++j) {
    int k = lane + 32 * j;
    uint16_t hi, lo; bf16_split(x[j] * rn, hi, lo);
    int kc = k >> 5, ko = k & 31, lp, jj;
    frag_slot(col, ko, lp, jj);
    size_t idx = (((size_t)apair * KC + kc) * 32 + lp) * 16 + jj;
    pk_hi[idx] = hi; pk_lo[idx] = lo;
  }
}

// ---- per-a mean of normalized descriptors (both sets) ------------------------
__global__ void ost_dmean(const float* __restrict__ ds, const float* __restrict__ dt,
                          const float* __restrict__ rn0, const float* __restrict__ rn1,
                          float* __restrict__ dm0, float* __restrict__ dm1) {
  int i = (int)(blockIdx.x * blockDim.x + threadIdx.x);
  if (i >= An * Dn) return;
  int a = i / Dn, d = i % Dn;
  float s0 = 0.f, s1 = 0.f;
#pragma unroll
  for (int n = 0; n < Nn; ++n) {
    s0 += ds[((size_t)a * Nn + n) * Dn + d] * rn0[a * Nn + n];
    s1 += dt[((size_t)a * Nn + n) * Dn + d] * rn1[a * Nn + n];
  }
  dm0[i] = s0 * (1.f / Nn);
  dm1[i] = s1 * (1.f / Nn);
}

// ---- pooled image features (mean over T of normalized rows) ------------------
__global__ void ost_img_pool(const float* __restrict__ imf,
                             const float* __restrict__ rn_img,
                             float* __restrict__ pool) {
  int i = (int)(blockIdx.x * blockDim.x + threadIdx.x);
  if (i >= Bn * Dn) return;
  int b = i / Dn, d = i % Dn;
  float s = 0.f;
#pragma unroll
  for (int t = 0; t < Tn; ++t)
    s += imf[((size_t)b * Tn + t) * Dn + d] * rn_img[b * Tn + t];
  pool[i] = s * (1.f / Tn);
}

// ---- pooled-logit term: initializes d_out = 0.5*ls*(pool·(dm_s+dm_t)) --------
__global__ void ost_pool_logits(const float* __restrict__ pool,
                                const float* __restrict__ dm0,
                                const float* __restrict__ dm1,
                                const float* __restrict__ logit_scale,
                                float* __restrict__ out) {
  int i = (int)(blockIdx.x * blockDim.x + threadIdx.x);
  if (i >= Bn * An) return;
  int b = i / An, a = i % An;
  float acc = 0.f;
  const float* pb = pool + (size_t)b * Dn;
  const float* d0 = dm0 + (size_t)a * Dn;
  const float* d1 = dm1 + (size_t)a * Dn;
  for (int d = 0; d < Dn; ++d) acc += pb[d] * (d0[d] + d1[d]);
  out[i] = 0.5f * __expf(logit_scale[0]) * acc;
}

// ---- fused WMMA GEMM + Sinkhorn + OT score -----------------------------------
// One block = 8 waves sharing b; A(b) fragments DMA'd to LDS by the TDM once per
// block; each wave owns one a-pair and streams B from global (L2-resident).
__global__ void __launch_bounds__(256)
ost_wmma_sinkhorn(const uint16_t* __restrict__ A_hi, const uint16_t* __restrict__ A_lo,
                  const uint16_t* __restrict__ B_hi0, const uint16_t* __restrict__ B_lo0,
                  const uint16_t* __restrict__ B_hi1, const uint16_t* __restrict__ B_lo1,
                  const float* __restrict__ logit_scale, float* __restrict__ out) {
  __shared__ uint16_t lds_a[2][TDM_ELEMS];  // 32KB: this b's A hi/lo fragments

  const int wave = (int)(threadIdx.x >> 5);
  const int lane = (int)(threadIdx.x & 31);
  const int b    = (int)blockIdx.x / (APAIR / 8);
  const int ap   = ((int)blockIdx.x % (APAIR / 8)) * 8 + wave;

  if (threadIdx.x < 32) {  // wave 0 drives the TDM (EXEC ignored by tensor ops)
    tdm_load_1d(A_hi + (size_t)b * TDM_ELEMS, (uint32_t)(uintptr_t)&lds_a[0][0]);
    tdm_load_1d(A_lo + (size_t)b * TDM_ELEMS, (uint32_t)(uintptr_t)&lds_a[1][0]);
    __builtin_amdgcn_s_wait_tensorcnt(0);
  }
  __syncthreads();

  const uint16_t* Bh[2] = { B_hi0, B_hi1 };
  const uint16_t* Bl[2] = { B_lo0, B_lo1 };
  const size_t bbase = ((size_t)ap * KC * 32 + lane) * 16;
  const uint16_t* la_hi = &lds_a[0][lane * 16];
  const uint16_t* la_lo = &lds_a[1][lane * 16];

  float score[2];
#pragma unroll
  for (int s = 0; s < 2; ++s) {
    v8f acc;
#pragma unroll
    for (int i = 0; i < 8; ++i) acc[i] = 0.f;

    for (int kc = 0; kc < KC; ++kc) {
      const v16bf a_hi = *(const v16bf*)(la_hi + kc * FRAG);   // ds_load_b128 x2
      const v16bf a_lo = *(const v16bf*)(la_lo + kc * FRAG);
      const v16bf b_hi = *(const v16bf*)(Bh[s] + bbase + (size_t)kc * FRAG);
      const v16bf b_lo = *(const v16bf*)(Bl[s] + bbase + (size_t)kc * FRAG);
      if (kc + 1 < KC) {  // pull next B chunk toward L1 (global_prefetch_b8)
        __builtin_prefetch(Bh[s] + bbase + (size_t)(kc + 1) * FRAG, 0, 3);
        __builtin_prefetch(Bl[s] + bbase + (size_t)(kc + 1) * FRAG, 0, 3);
      }
      acc = __builtin_amdgcn_wmma_f32_16x16x32_bf16(false, a_hi, false, b_hi,
                                                    (short)0, acc, false, false);
      acc = __builtin_amdgcn_wmma_f32_16x16x32_bf16(false, a_hi, false, b_lo,
                                                    (short)0, acc, false, false);
      acc = __builtin_amdgcn_wmma_f32_16x16x32_bf16(false, a_lo, false, b_hi,
                                                    (short)0, acc, false, false);
    }

    // acc[i] = sim[t][n], t = (lane/16)*8 + i, n = lane%16 -> (pair = n/8, col = n%8)
    float Km[8], r[8], r0[8];
#pragma unroll
    for (int i = 0; i < 8; ++i) { Km[i] = __expf((acc[i] - 1.f) * INV_EPS); r[i] = 1.f; }
    float c = 1.f;
    const float u = 1.f / Tn, vv = 1.f / Nn;
    float err = 1e30f;

    for (int it = 0; it < MAX_ITER && err >= THRESH; ++it) {
      float kcv[8];
#pragma unroll
      for (int i = 0; i < 8; ++i) { r0[i] = r[i]; kcv[i] = Km[i] * c; }
      // sum over the pair's 8 columns: xor-1/2/4 butterflies stay inside the
      // 8-lane group -> pure VALU dpp8, co-executes with the XDL pipe
#pragma unroll
      for (int i = 0; i < 8; ++i) {
        kcv[i] = XOR_ADD(kcv[i], DPP8_XOR1);
        kcv[i] = XOR_ADD(kcv[i], DPP8_XOR2);
        kcv[i] = XOR_ADD(kcv[i], DPP8_XOR4);
      }
      float sKr = 0.f, e = 0.f;
#pragma unroll
      for (int i = 0; i < 8; ++i) {
        r[i] = u * __builtin_amdgcn_rcpf(kcv[i]);   // v_rcp_f32, no div fixup
        sKr += Km[i] * r[i];
        e   += fabsf(r[i] - r0[i]);
      }
      sKr += __shfl_xor(sKr, 16, 32);  // add the other 8 t's (other lane half)
      c = vv * __builtin_amdgcn_rcpf(sKr);
      e += __shfl_xor(e, 16, 32);      // |dr| summed over all 16 t's
      e *= (1.f / 16.f);
      e = fmaxf(e, __shfl_xor(e, 8, 32));  // wave-uniform max over both pairs
      err = e;
    }

    // score_ot = sum_{t,n} r[t] * c[n] * K[t][n] * sim[t][n]
    float sc = 0.f;
#pragma unroll
    for (int i = 0; i < 8; ++i) sc += r[i] * Km[i] * acc[i];
    sc *= c;
    sc = XOR_ADD(sc, DPP8_XOR1);
    sc = XOR_ADD(sc, DPP8_XOR2);
    sc = XOR_ADD(sc, DPP8_XOR4);
    sc += __shfl_xor(sc, 16, 32);
    score[s] = sc;
  }

  float ls = __expf(logit_scale[0]);
  if (lane < 16 && (lane & 7) == 0) {  // lane 0 -> pair 0, lane 8 -> pair 1
    int a = ap * 2 + (lane >> 3);
    out[(size_t)b * An + a] += 0.5f * ls * (score[0] + score[1]);
  }
}

extern "C" void kernel_launch(void* const* d_in, const int* in_sizes, int n_in,
                              void* d_out, int out_size, void* d_ws, size_t ws_size,
                              hipStream_t stream) {
  const float* imf = (const float*)d_in[0];   // (B,T,D)
  const float* ds  = (const float*)d_in[1];   // (A,N,D) spatial
  const float* dt  = (const float*)d_in[2];   // (A,N,D) temporal
  const float* lsc = (const float*)d_in[3];   // (1,)
  float* out = (float*)d_out;                 // (B,A)

  char* ws = (char*)d_ws;
  size_t off = 0;
  auto take = [&](size_t bytes) -> void* {
    void* p = ws + off;
    off += (bytes + 255) & ~(size_t)255;
    return p;
  };
  const size_t afrag = (size_t)Bn    * KC * FRAG * sizeof(uint16_t);  // 4 MB
  const size_t bfrag = (size_t)APAIR * KC * FRAG * sizeof(uint16_t);  // 3.2 MB
  uint16_t* A_hi  = (uint16_t*)take(afrag);
  uint16_t* A_lo  = (uint16_t*)take(afrag);
  uint16_t* B_hi0 = (uint16_t*)take(bfrag);
  uint16_t* B_lo0 = (uint16_t*)take(bfrag);
  uint16_t* B_hi1 = (uint16_t*)take(bfrag);
  uint16_t* B_lo1 = (uint16_t*)take(bfrag);
  float* rn_img = (float*)take((size_t)Bn * Tn * sizeof(float));
  float* rn_ds  = (float*)take((size_t)An * Nn * sizeof(float));
  float* rn_dt  = (float*)take((size_t)An * Nn * sizeof(float));
  float* pool   = (float*)take((size_t)Bn * Dn * sizeof(float));
  float* dm0    = (float*)take((size_t)An * Dn * sizeof(float));
  float* dm1    = (float*)take((size_t)An * Dn * sizeof(float));

  // 1) normalize + pre-swizzle into WMMA fragment layout
  ost_norm_pack_img<<<(Bn * Tn) / 8, 256, 0, stream>>>(imf, A_hi, A_lo, rn_img);
  ost_norm_pack_desc<<<(An * Nn) / 8, 256, 0, stream>>>(ds, B_hi0, B_lo0, rn_ds);
  ost_norm_pack_desc<<<(An * Nn) / 8, 256, 0, stream>>>(dt, B_hi1, B_lo1, rn_dt);

  // 2) pooled terms
  ost_dmean<<<(An * Dn + 255) / 256, 256, 0, stream>>>(ds, dt, rn_ds, rn_dt, dm0, dm1);
  ost_img_pool<<<(Bn * Dn + 255) / 256, 256, 0, stream>>>(imf, rn_img, pool);
  ost_pool_logits<<<(Bn * An + 255) / 256, 256, 0, stream>>>(pool, dm0, dm1, lsc, out);

  // 3) fused WMMA similarity GEMM + Sinkhorn OT, accumulates into out
  //    grid: Bn * (APAIR/8) blocks; each block = one b, 8 a-pairs
  ost_wmma_sinkhorn<<<Bn * (APAIR / 8), 256, 0, stream>>>(
      A_hi, A_lo, B_hi0, B_lo0, B_hi1, B_lo1, lsc, out);
}
